// proj_net_90340342104673
// MI455X (gfx1250) — matrix operand order repaired and checked by reading, not compiled, not run on hardware
//
#include <hip/hip_runtime.h>
#include <hip/hip_bf16.h>
#include <stdint.h>

// ---------------- problem sizes (match reference) ----------------
#define HID     1024
#define BATCH   128
#define TSTEPS  512
#define INDIM   2

// ---------------- mapping ----------------
#define NWG         16                 // max cluster size; 1 WGP each
#define COLS_PER_WG (HID / NWG)        // 64 N-columns per workgroup
#define THREADS     256                // 8 wave32
#define NWAVES      8
#define KCHUNK      128                // TDM-staged K chunk of h (f16)
#define NCHUNK      (HID / KCHUNK)     // 8

typedef _Float16 half_t;
typedef __attribute__((ext_vector_type(16))) _Float16 v16h;
typedef __attribute__((ext_vector_type(8)))  _Float16 v8h;
typedef __attribute__((ext_vector_type(8)))  float    v8f;
typedef __attribute__((ext_vector_type(4)))  unsigned v4u;
typedef __attribute__((ext_vector_type(8)))  unsigned v8u;

// ws layout: h16 ping[128*1024] | h16 pong[128*1024] | barrier counter
#define WS_HBUF_HALFS   (BATCH * HID)
#define WS_BAR_U32      (2 * WS_HBUF_HALFS / 2)   // byte offset 512KB -> u32 index 131072

// ---------------- init: zero h0 and the barrier counter ----------------
__global__ void init_ws_kernel(unsigned* __restrict__ ws) {
    const int n_u32 = WS_HBUF_HALFS / 2;           // buffer 0 as u32
    for (int i = blockIdx.x * blockDim.x + threadIdx.x; i < n_u32;
         i += gridDim.x * blockDim.x)
        ws[i] = 0u;
    if (blockIdx.x == 0 && threadIdx.x == 0) ws[WS_BAR_U32] = 0u;
}

// ---------------- device-wide barrier: cluster barrier + monotonic atomic ----------------
__device__ __forceinline__ void device_barrier(unsigned* bar, unsigned target) {
    __builtin_amdgcn_fence(__ATOMIC_RELEASE, "agent");
    __syncthreads();
    __builtin_amdgcn_s_cluster_barrier();          // real sync when dispatched as a cluster; NOP otherwise
    if (threadIdx.x == 0) {
        __hip_atomic_fetch_add(bar, 1u, __ATOMIC_RELEASE, __HIP_MEMORY_SCOPE_AGENT);
        while (__hip_atomic_load(bar, __ATOMIC_ACQUIRE, __HIP_MEMORY_SCOPE_AGENT) < target)
            __builtin_amdgcn_s_sleep(1);
    }
    __syncthreads();
    __builtin_amdgcn_fence(__ATOMIC_ACQUIRE, "agent");
}

// ---------------- TDM: async copy of a 16 x KCHUNK f16 tile (row stride HID) into LDS ----------------
// D# per CDNA5 ISA 8.3/8.4: group0 = {count|flags, lds_addr, global_addr lo, global_addr hi | type=2}
// group1 = {mask/data_size, tensor_dim0, tensor_dim1, tile_dim0, tile_dim1, dim0_stride, 0, 0}
__device__ __forceinline__ void tdm_load_htile(const half_t* gsrc, const half_t* lds_dst) {
    unsigned long long ga = (unsigned long long)(uintptr_t)gsrc;
    unsigned lds_off = (unsigned)(uintptr_t)lds_dst;            // low 32 bits of LDS aperture = LDS byte addr
    // force wave-uniform scalars for the SGPR descriptor
    unsigned u_lds = (unsigned)__builtin_amdgcn_readfirstlane((int)lds_off);
    unsigned u_glo = (unsigned)__builtin_amdgcn_readfirstlane((int)(unsigned)ga);
    unsigned u_ghi = (unsigned)__builtin_amdgcn_readfirstlane((int)(unsigned)(ga >> 32));
    v4u g0 = { 1u,                                              // count=1 valid descriptor
               u_lds,
               u_glo,
               (u_ghi & 0x01FFFFFFu) | 0x80000000u };           // addr[56:32] | type=2 ("image")
    v8u g1 = { (1u << 16),                                      // workgroup_mask=0, data_size=1 (2 bytes)
               ((unsigned)HID) << 16,                           // tensor_dim0 = 1024 (lo16 at bit 48)
               ((unsigned)BATCH) << 16,                         // tensor_dim1 = 128  (lo16 at bit 80)
               ((unsigned)KCHUNK) << 16,                        // tile_dim0 = 128    (bits 127:112)
               16u,                                             // tile_dim1 = 16     (bits 143:128)
               (unsigned)HID,                                   // tensor_dim0_stride = 1024 elements
               0u, 0u };
    v4u gz = { 0u, 0u, 0u, 0u };
    asm volatile("tensor_load_to_lds %0, %1, %2, %3"
                 :: "s"(g0), "s"(g1), "s"(gz), "s"(gz)
                 : "memory");
}

// ---------------- WMMA fragment gathers from LDS ----------------
// A (16x32 f16, M x K): lane L<16 -> M=L, K = kk+{0..7, 16..23}; lane>=16 -> K = kk+{8..15, 24..31}
__device__ __forceinline__ v16h ld_frag_a(const half_t* abuf, int lane, int kk) {
    int r    = lane & 15;
    int koff = kk + ((lane & 16) ? 8 : 0);
    const v8h* p0 = (const v8h*)(abuf + r * KCHUNK + koff);
    const v8h* p1 = (const v8h*)(abuf + r * KCHUNK + koff + 16);
    v8h lo = *p0, hi = *p1;
    return __builtin_shufflevector(lo, hi, 0,1,2,3,4,5,6,7,8,9,10,11,12,13,14,15);
}
// B (32x16 f16, K x N): lane L -> N=L%16; lanes 0-15 hold K=k+{0..7, 16..23}, lanes 16-31 hold K=k+{8..15, 24..31}
__device__ __forceinline__ v16h ld_frag_b(const half_t* wlds, int lane, int nrow0, int k) {
    int n    = nrow0 + (lane & 15);
    int koff = k + ((lane & 16) ? 8 : 0);
    const v8h* p0 = (const v8h*)(wlds + (size_t)n * HID + koff);
    v8h lo = p0[0], hi = p0[2];
    return __builtin_shufflevector(lo, hi, 0,1,2,3,4,5,6,7,8,9,10,11,12,13,14,15);
}

// ---------------- persistent RNN kernel ----------------
__global__ __launch_bounds__(THREADS, 1)
void rnn_persistent_kernel(const float* __restrict__ inputs,   // [B, T, 2]
                           const float* __restrict__ W_in,     // [H, 2]
                           const float* __restrict__ W_rec,    // [H, H]
                           const float* __restrict__ W_out,    // [1, H]
                           const float* __restrict__ b_out,    // [1]
                           float* __restrict__ out,            // [B]
                           half_t* __restrict__ h16,           // ws: [2][B][H] f16 ping-pong
                           unsigned* __restrict__ bar) {
    extern __shared__ char smem[];
    half_t* wlds = (half_t*)smem;                               // [64][1024] f16 = 128 KB (resident W slice)
    half_t* alds = (half_t*)(smem + COLS_PER_WG * HID * 2);     // 8 waves * 2 bufs * 16*128 f16 = 64 KB
    float*  xin  = (float*)(smem + COLS_PER_WG * HID * 2 + NWAVES * 2 * 16 * KCHUNK * 2); // [128][2]

    const int wg    = blockIdx.x;
    const int tid   = threadIdx.x;
    const int wave  = tid >> 5;
    const int lane  = tid & 31;
    const int nbase = wg * COLS_PER_WG;

    // ---- one-time: stage this WG's 64-row slice of W_rec into LDS as f16 ----
    for (int idx = tid; idx < COLS_PER_WG * (HID / 4); idx += THREADS) {
        int r  = idx / (HID / 4);
        int k4 = idx % (HID / 4);
        float4 v = ((const float4*)(W_rec + (size_t)(nbase + r) * HID))[k4];
        half_t* d = wlds + (size_t)r * HID + k4 * 4;
        d[0] = (half_t)v.x; d[1] = (half_t)v.y; d[2] = (half_t)v.z; d[3] = (half_t)v.w;
    }
    // per-lane input-projection weights for the 4 owned column tiles (constant over time)
    float win0[4], win1[4];
#pragma unroll
    for (int c = 0; c < 4; ++c) {
        int n = nbase + c * 16 + (lane & 15);
        win0[c] = W_in[n * 2 + 0];
        win1[c] = W_in[n * 2 + 1];
    }
    __syncthreads();

    half_t* abuf0 = alds + (size_t)wave * (2 * 16 * KCHUNK);
    half_t* abuf1 = abuf0 + 16 * KCHUNK;
    const v8f vzero = {0.f,0.f,0.f,0.f,0.f,0.f,0.f,0.f};

    unsigned target = NWG;
    for (int t = 0; t < TSTEPS; ++t) {
        const half_t* h_r = h16 + (size_t)(t & 1) * WS_HBUF_HALFS;
        half_t*       h_w = h16 + (size_t)((t + 1) & 1) * WS_HBUF_HALFS;

        // stage x_t (128 rows x 2) into LDS
        {
            int b = tid >> 1, i = tid & 1;
            if (b < BATCH) xin[tid] = inputs[((size_t)b * TSTEPS + t) * INDIM + i];
        }
        __syncthreads();

        v8f acc[4] = { vzero, vzero, vzero, vzero };

        // K loop: TDM double-buffered staging of this wave's 16 rows of h
        const half_t* hrow = h_r + (size_t)(wave * 16) * HID;
        tdm_load_htile(hrow, abuf0);
        for (int kc = 0; kc < NCHUNK; ++kc) {
            if (kc + 1 < NCHUNK) {
                tdm_load_htile(hrow + (kc + 1) * KCHUNK, ((kc + 1) & 1) ? abuf1 : abuf0);
                __builtin_amdgcn_s_wait_tensorcnt(1);   // current chunk landed (in-order TDM)
            } else {
                __builtin_amdgcn_s_wait_tensorcnt(0);
            }
            const half_t* ab = (kc & 1) ? abuf1 : abuf0;
#pragma unroll
            for (int kk = 0; kk < KCHUNK; kk += 32) {
                v16h aF = ld_frag_a(ab, lane, kk);
#pragma unroll
                for (int c = 0; c < 4; ++c) {
                    v16h bF = ld_frag_b(wlds, lane, c * 16, kc * KCHUNK + kk);
                    acc[c] = __builtin_amdgcn_wmma_f32_16x16x32_f16(
                        false, aF, false, bF, (short)0, acc[c], false, false);
                }
            }
        }

        // epilogue: h = relu(acc + W_in x_t); store f16 to ping-pong buffer
        {
            int rbase = wave * 16 + ((lane & 16) ? 8 : 0);
            int nlo   = lane & 15;
#pragma unroll
            for (int v = 0; v < 8; ++v) {
                int   m  = rbase + v;
                float x0 = xin[m * 2 + 0];
                float x1 = xin[m * 2 + 1];
#pragma unroll
                for (int c = 0; c < 4; ++c) {
                    float h = acc[c][v] + x0 * win0[c] + x1 * win1[c];
                    h = fmaxf(h, 0.0f);
                    int n = nbase + c * 16 + nlo;
                    h_w[(size_t)m * HID + n] = (half_t)h;
                }
            }
        }

        device_barrier(bar, target);
        target += NWG;
    }

    // ---- output layer: out[b] = h_last . W_out + b_out (h_last is in buffer 0: T is even) ----
    if (wg == 0) {
        const half_t* hl = h16;   // buffer (TSTEPS & 1) == 0
        for (int b = tid; b < BATCH; b += THREADS) {
            float s = 0.0f;
            for (int n = 0; n < HID; ++n)
                s = fmaf((float)hl[(size_t)b * HID + n], W_out[n], s);
            out[b] = s + b_out[0];
        }
    }
}

// ---------------- launcher ----------------
extern "C" void kernel_launch(void* const* d_in, const int* in_sizes, int n_in,
                              void* d_out, int out_size, void* d_ws, size_t ws_size,
                              hipStream_t stream) {
    (void)in_sizes; (void)n_in; (void)out_size; (void)ws_size;
    const float* inputs = (const float*)d_in[0];
    const float* W_in   = (const float*)d_in[1];
    const float* W_rec  = (const float*)d_in[2];
    const float* W_out  = (const float*)d_in[3];
    const float* b_out  = (const float*)d_in[4];
    float*       out    = (float*)d_out;

    half_t*   h16 = (half_t*)d_ws;
    unsigned* bar = (unsigned*)d_ws + WS_BAR_U32;

    init_ws_kernel<<<64, 256, 0, stream>>>((unsigned*)d_ws);

    size_t smem = (size_t)COLS_PER_WG * HID * 2      // W slice f16
                + (size_t)NWAVES * 2 * 16 * KCHUNK * 2 // A double buffers
                + (size_t)BATCH * INDIM * 4;           // x_t staging
    (void)hipFuncSetAttribute((const void*)rnn_persistent_kernel,
                              hipFuncAttributeMaxDynamicSharedMemorySize, (int)smem);
    rnn_persistent_kernel<<<NWG, THREADS, smem, stream>>>(
        inputs, W_in, W_rec, W_out, b_out, out, h16, bar);
}
